// HeteroHAN_8495445311833
// MI455X (gfx1250) — matrix operand order repaired and checked by reading, not compiled, not run on hardware
//
#include <hip/hip_runtime.h>
#include <hip/hip_bf16.h>
#include <math.h>

// ---------------------------------------------------------------------------
// HeteroHAN (2-layer hetero GAT) for MI455X / gfx1250, wave32 + WMMA bf16.
//   - Projections: C[M,256] = A_bf16[M,256] x W[256,256] via
//     v_wmma_f32_16x16x32_bf16 (fp32 accumulate). W pre-transposed to Wt so
//     B fragments are contiguous 32B per lane.
//   - Edge softmax: ordered-uint atomicMax pass for segment max, then a
//     single scatter pass accumulating ex*fs[src] and den with HW f32 atomics.
//   - out = (den>0 ? accum/den : 0) + b, elu, written per relation.
//   - Row guards scalarized (readfirstlane) so WMMA always executes with
//     EXEC all-1s under scalar branches, per ISA 7.12 restrictions.
// ---------------------------------------------------------------------------

#define NNODES 50000
#define NEDGES 500000
#define NHEADS 8
#define DFEAT  256
#define SLOPE_ 0.2f

typedef __attribute__((ext_vector_type(16))) __bf16 v16bf;
typedef __attribute__((ext_vector_type(8)))  __bf16 v8bf;
typedef __attribute__((ext_vector_type(8)))  float  v8f;

// ---------------- WMMA fragment helpers (CDNA5 16x16x32 bf16 layouts) -------

// A (16x32, MxK): lane L holds row m = L&15; K chunks [kb,kb+8) and
// [kb+8,kb+16) with kb = k0 + (L>>4)*16  (two contiguous 16B loads).
__device__ __forceinline__ v16bf load_a_frag(const __bf16* A, int row0, int k0, int lane) {
  const int m  = row0 + (lane & 15);
  const int kb = k0 + ((lane >> 4) << 4);
  const __bf16* base = A + (size_t)m * DFEAT + kb;
  v8bf lo = *(const v8bf*)(base);
  v8bf hi = *(const v8bf*)(base + 8);
  v16bf r;
#pragma unroll
  for (int i = 0; i < 8; ++i) { r[i] = lo[i]; r[8 + i] = hi[i]; }
  return r;
}

// B (32x16, KxN): lane L holds col n = L&15, K = k0+(L>>4)*16 .. +15 packed
// 2-per-VGPR. With Wt = W^T row-major this is one contiguous 32B load.
__device__ __forceinline__ v16bf load_b_frag(const __bf16* Wt, int col0, int k0, int lane) {
  const int n  = col0 + (lane & 15);
  const int kb = k0 + ((lane >> 4) << 4);
  return *(const v16bf*)(Wt + (size_t)n * DFEAT + kb);
}

// C/D f32 16x16: VGPR r -> row (row0 + (lane>>4)*8 + r), col (col0 + lane&15).
__device__ __forceinline__ void store_c_tile(float* C, int row0, int col0, int lane, v8f acc) {
  const int n  = col0 + (lane & 15);
  const int r0 = row0 + ((lane >> 4) << 3);
#pragma unroll
  for (int r = 0; r < 8; ++r) C[(size_t)(r0 + r) * DFEAT + n] = acc[r];
}

// C[M x 256] = A(bf16)[M x 256] * W, Wt = W^T (bf16 [256 x 256] row-major).
// Block: 256 threads = 8 waves; wave tile 32x32 (4 WMMA accums); block 128x64.
__global__ __launch_bounds__(256) void gemm_bf16_wmma_kernel(
    const __bf16* __restrict__ A, const __bf16* __restrict__ Wt,
    float* __restrict__ C, int M) {
  const int lane = threadIdx.x & 31;
  const int wave = threadIdx.x >> 5;
  const int wm = wave & 3;   // 4 waves along M
  const int wn = wave >> 2;  // 2 waves along N
  const int rowBase = blockIdx.x * 128 + wm * 32;
  const int colBase = blockIdx.y * 64 + wn * 32;
  // Wave-uniform guards (M % 16 == 0); force into SGPRs so the branches are
  // scalar and WMMA never executes inside an EXEC-masked region.
  const int ok0 = __builtin_amdgcn_readfirstlane(((rowBase + 16) <= M) ? 1 : 0);
  const int ok1 = __builtin_amdgcn_readfirstlane(((rowBase + 32) <= M) ? 1 : 0);
  if (!ok0) return;

  v8f acc00 = {}, acc01 = {}, acc10 = {}, acc11 = {};
#pragma unroll
  for (int k0 = 0; k0 < DFEAT; k0 += 32) {
    v16bf b0 = load_b_frag(Wt, colBase,      k0, lane);
    v16bf b1 = load_b_frag(Wt, colBase + 16, k0, lane);
    v16bf a0 = load_a_frag(A, rowBase, k0, lane);
    acc00 = __builtin_amdgcn_wmma_f32_16x16x32_bf16(false, a0, false, b0, (short)0, acc00, false, false);
    acc01 = __builtin_amdgcn_wmma_f32_16x16x32_bf16(false, a0, false, b1, (short)0, acc01, false, false);
    if (ok1) {
      v16bf a1 = load_a_frag(A, rowBase + 16, k0, lane);
      acc10 = __builtin_amdgcn_wmma_f32_16x16x32_bf16(false, a1, false, b0, (short)0, acc10, false, false);
      acc11 = __builtin_amdgcn_wmma_f32_16x16x32_bf16(false, a1, false, b1, (short)0, acc11, false, false);
    }
  }
  store_c_tile(C, rowBase, colBase,      lane, acc00);
  store_c_tile(C, rowBase, colBase + 16, lane, acc01);
  if (ok1) {
    store_c_tile(C, rowBase + 16, colBase,      lane, acc10);
    store_c_tile(C, rowBase + 16, colBase + 16, lane, acc11);
  }
}

// ---------------- elementwise / edge kernels --------------------------------

__global__ void convert_bf16_kernel(const float* __restrict__ in, __bf16* __restrict__ out,
                                    float scale, int n) {
  int i = blockIdx.x * blockDim.x + threadIdx.x;
  if (i < n) out[i] = (__bf16)(in[i] * scale);
}

__global__ void wtrans_kernel(const float* __restrict__ W, __bf16* __restrict__ Wt) {
  int gid = blockIdx.x * blockDim.x + threadIdx.x;  // 65536 threads
  int n = gid & 255, k = gid >> 8;
  Wt[(size_t)n * DFEAT + k] = (__bf16)W[(size_t)k * DFEAT + n];
}

// el/er: out[i,h] = sum_d f[i, h*32+d] * a[h*32+d]
__global__ void attn_proj_kernel(const float* __restrict__ f, const float* __restrict__ a,
                                 float* __restrict__ out, int n) {
  int gid = blockIdx.x * blockDim.x + threadIdx.x;
  if (gid >= n) return;
  int i = gid >> 3, h = gid & 7;
  const float* fr = f + (size_t)i * DFEAT + h * 32;
  const float* ar = a + h * 32;
  float s = 0.f;
#pragma unroll
  for (int d = 0; d < 32; ++d) s += fr[d] * ar[d];
  out[gid] = s;
}

__global__ void fill_u32_kernel(unsigned* __restrict__ p, unsigned v, int n) {
  int i = blockIdx.x * blockDim.x + threadIdx.x;
  if (i < n) p[i] = v;
}
__global__ void fill_f32_kernel(float* __restrict__ p, float v, int n) {
  int i = blockIdx.x * blockDim.x + threadIdx.x;
  if (i < n) p[i] = v;
}

// ordered-uint encoding: monotone map f32 -> u32 so unsigned max == float max
__global__ void edge_max_kernel(const int* __restrict__ s, const int* __restrict__ d,
                                const float* __restrict__ el, const float* __restrict__ er,
                                unsigned* __restrict__ emax) {
  int e = blockIdx.x * blockDim.x + threadIdx.x;
  if (e >= NEDGES) return;
  int si = s[e], di = d[e];
#pragma unroll
  for (int h = 0; h < NHEADS; ++h) {
    float x = el[si * 8 + h] + er[di * 8 + h];
    x = x > 0.f ? x : SLOPE_ * x;
    unsigned u = __float_as_uint(x);
    unsigned k = (u & 0x80000000u) ? ~u : (u | 0x80000000u);
    atomicMax(&emax[di * 8 + h], k);
  }
}

// decode in place (uint key -> float max); untouched slots (key 0) -> NaN -> 0
__global__ void decode_max_kernel(unsigned* __restrict__ buf, int n) {
  int i = blockIdx.x * blockDim.x + threadIdx.x;
  if (i >= n) return;
  unsigned k = buf[i];
  unsigned u = (k & 0x80000000u) ? (k ^ 0x80000000u) : ~k;
  float f = __uint_as_float(u);
  if (!isfinite(f)) f = 0.f;
  ((float*)buf)[i] = f;
}

// One thread per (edge, 4-channel chunk): 64 threads/edge.
// accum[dst] += ex * fs[src]; den[dst,h] += ex (once per head).
__global__ void edge_accum_kernel(const int* __restrict__ s, const int* __restrict__ d,
                                  const float* __restrict__ el, const float* __restrict__ er,
                                  const float* __restrict__ m, const float* __restrict__ fs,
                                  float* __restrict__ den, float* __restrict__ accum) {
  int gid = blockIdx.x * blockDim.x + threadIdx.x;  // NEDGES*64 = 32M
  if (gid >= NEDGES * 64) return;
  int e = gid >> 6;
  int j = gid & 63;
  int h = j >> 3;
  int c0 = h * 32 + (j & 7) * 4;
  int si = s[e], di = d[e];
  float x = el[si * 8 + h] + er[di * 8 + h];
  x = x > 0.f ? x : SLOPE_ * x;
  float ex = __expf(x - m[di * 8 + h]);
  if ((j & 7) == 0) unsafeAtomicAdd(&den[di * 8 + h], ex);
  float4 fv = *(const float4*)(fs + (size_t)si * DFEAT + c0);
  float* ar = accum + (size_t)di * DFEAT + c0;
  unsafeAtomicAdd(ar + 0, ex * fv.x);
  unsafeAtomicAdd(ar + 1, ex * fv.y);
  unsafeAtomicAdd(ar + 2, ex * fv.z);
  unsafeAtomicAdd(ar + 3, ex * fv.w);
}

// out = elu((den>0 ? accum/den : 0) + b); assign or add into strided target.
__global__ void finalize_kernel(const float* __restrict__ accum, const float* __restrict__ den,
                                const float* __restrict__ b, float* __restrict__ out,
                                int ldout, int colOff, int addMode) {
  int gid = blockIdx.x * blockDim.x + threadIdx.x;  // NNODES*256
  if (gid >= NNODES * DFEAT) return;
  int i = gid >> 8, c = gid & 255, h = c >> 5;
  float dn = den[i * 8 + h];
  float v = (dn > 0.f ? accum[gid] / dn : 0.f) + b[c];
  v = v > 0.f ? v : expm1f(v);
  float* o = out + (size_t)i * ldout + colOff + c;
  if (addMode) *o += v; else *o = v;
}

// ---------------------------------------------------------------------------

extern "C" void kernel_launch(void* const* d_in, const int* in_sizes, int n_in,
                              void* d_out, int out_size, void* d_ws, size_t ws_size,
                              hipStream_t stream) {
  (void)in_sizes; (void)n_in; (void)out_size; (void)ws_size;
  const float* xm  = (const float*)d_in[0];
  const float* xt  = (const float*)d_in[1];
  const int*   src = (const int*)d_in[2];   // [4, E]
  const int*   dst = (const int*)d_in[3];   // [4, E]
  const float* W1  = (const float*)d_in[4]; // [4, 256, 256]
  const float* a1l = (const float*)d_in[5]; // [4, 8, 32]
  const float* a1r = (const float*)d_in[6];
  const float* b1  = (const float*)d_in[7];
  const float* W2  = (const float*)d_in[8];
  const float* a2l = (const float*)d_in[9];
  const float* a2r = (const float*)d_in[10];
  const float* b2  = (const float*)d_in[11];
  float* out = (float*)d_out;               // [N, 4*256]

  // ---- workspace carve-out (~315 MB) ----
  char* ws = (char*)d_ws;
  size_t off = 0;
  auto take = [&](size_t bytes) -> void* {
    void* p = ws + off;
    off += (bytes + 255) & ~(size_t)255;
    return p;
  };
  const size_t NF = (size_t)NNODES * DFEAT;
  __bf16* bfA   = (__bf16*)take(NF * 2);            // merchant-side bf16 feats
  __bf16* bfB   = (__bf16*)take(NF * 2);            // tobacco-side bf16 feats
  __bf16* Wt1   = (__bf16*)take((size_t)4 * 65536 * 2);
  __bf16* Wt2   = (__bf16*)take((size_t)4 * 65536 * 2);
  float*  fs    = (float*)take(NF * 4);
  float*  fd    = (float*)take(NF * 4);
  float*  accum = (float*)take(NF * 4);
  float*  hmsum = (float*)take(NF * 4);
  float*  htbuf = (float*)take(NF * 4);
  float*  el    = (float*)take((size_t)NNODES * 8 * 4);
  float*  er    = (float*)take((size_t)NNODES * 8 * 4);
  float*  den   = (float*)take((size_t)NNODES * 8 * 4);
  unsigned* emax = (unsigned*)take((size_t)NNODES * 8 * 4);

  const dim3 gemmGrid((NNODES + 127) / 128, DFEAT / 64);
  auto gemm = [&](const __bf16* Ain, const __bf16* Wt, float* Cout) {
    gemm_bf16_wmma_kernel<<<gemmGrid, 256, 0, stream>>>(Ain, Wt, Cout, NNODES);
  };
  const int nNH  = NNODES * NHEADS;
  const int bNH  = (nNH + 255) / 256;
  const int bNF  = (int)((NF + 255) / 256);
  const int bE   = (NEDGES + 255) / 256;
  const int bE64 = (NEDGES * 64 + 255) / 256;

  auto relation = [&](const __bf16* srcFeat, const __bf16* dstFeat /*null == same*/,
                      const __bf16* Wt, const float* al, const float* ar, const float* bb,
                      const int* sE, const int* dE,
                      float* outp, int ldout, int colOff, int addMode) {
    gemm(srcFeat, Wt, fs);
    const float* fdp = fs;
    if (dstFeat) { gemm(dstFeat, Wt, fd); fdp = fd; }
    attn_proj_kernel<<<bNH, 256, 0, stream>>>(fs,  al, el, nNH);
    attn_proj_kernel<<<bNH, 256, 0, stream>>>(fdp, ar, er, nNH);
    fill_u32_kernel<<<bNH, 256, 0, stream>>>(emax, 0u, nNH);
    fill_f32_kernel<<<bNH, 256, 0, stream>>>(den, 0.f, nNH);
    fill_f32_kernel<<<bNF, 256, 0, stream>>>(accum, 0.f, (int)NF);
    edge_max_kernel<<<bE, 256, 0, stream>>>(sE, dE, el, er, emax);
    decode_max_kernel<<<bNH, 256, 0, stream>>>(emax, nNH);
    edge_accum_kernel<<<bE64, 256, 0, stream>>>(sE, dE, el, er, (const float*)emax, fs, den, accum);
    finalize_kernel<<<bNF, 256, 0, stream>>>(accum, den, bb, outp, ldout, colOff, addMode);
  };

  // ---- phase 0: bf16 inputs + transposed bf16 weights ----
  convert_bf16_kernel<<<bNF, 256, 0, stream>>>(xm, bfA, 1.f, (int)NF);
  convert_bf16_kernel<<<bNF, 256, 0, stream>>>(xt, bfB, 1.f, (int)NF);
  for (int r = 0; r < 4; ++r) {
    wtrans_kernel<<<256, 256, 0, stream>>>(W1 + (size_t)r * 65536, Wt1 + (size_t)r * 65536);
    wtrans_kernel<<<256, 256, 0, stream>>>(W2 + (size_t)r * 65536, Wt2 + (size_t)r * 65536);
  }

  // ---- layer 1 (elu per relation; merchant = mean of rels 1..3) ----
  // rel0 orders m->t ; rel1 geos m->m ; rel2 similar m->m ; rel3 rev t->m
  relation(bfA, bfB,     Wt1 + 0 * 65536, a1l + 0,   a1r + 0,   b1 + 0,
           src + 0 * NEDGES, dst + 0 * NEDGES, htbuf, 256, 0, 0);
  relation(bfA, nullptr, Wt1 + 1 * 65536, a1l + 256, a1r + 256, b1 + 256,
           src + 1 * NEDGES, dst + 1 * NEDGES, hmsum, 256, 0, 0);
  relation(bfA, nullptr, Wt1 + 2 * 65536, a1l + 512, a1r + 512, b1 + 512,
           src + 2 * NEDGES, dst + 2 * NEDGES, hmsum, 256, 0, 1);
  relation(bfB, bfA,     Wt1 + 3 * 65536, a1l + 768, a1r + 768, b1 + 768,
           src + 3 * NEDGES, dst + 3 * NEDGES, hmsum, 256, 0, 1);

  // hm = hmsum/3, ht = htbuf  -> bf16 inputs for layer 2
  convert_bf16_kernel<<<bNF, 256, 0, stream>>>(hmsum, bfA, 1.f / 3.f, (int)NF);
  convert_bf16_kernel<<<bNF, 256, 0, stream>>>(htbuf, bfB, 1.f, (int)NF);

  // ---- layer 2: per-relation slices of d_out [N, 4*256], final elu ----
  relation(bfA, bfB,     Wt2 + 0 * 65536, a2l + 0,   a2r + 0,   b2 + 0,
           src + 0 * NEDGES, dst + 0 * NEDGES, out, 1024, 0,   0);
  relation(bfA, nullptr, Wt2 + 1 * 65536, a2l + 256, a2r + 256, b2 + 256,
           src + 1 * NEDGES, dst + 1 * NEDGES, out, 1024, 256, 0);
  relation(bfA, nullptr, Wt2 + 2 * 65536, a2l + 512, a2r + 512, b2 + 512,
           src + 2 * NEDGES, dst + 2 * NEDGES, out, 1024, 512, 0);
  relation(bfB, bfA,     Wt2 + 3 * 65536, a2l + 768, a2r + 768, b2 + 768,
           src + 3 * NEDGES, dst + 3 * NEDGES, out, 1024, 768, 0);
}
